// AFT_Full_52733608460654
// MI455X (gfx1250) — matrix operand order repaired and checked by reading, not compiled, not run on hardware
//
#include <hip/hip_runtime.h>
#include <hip/hip_bf16.h>

// ---------------------------------------------------------------------------
// AFT-Full (B=8, S=D=1024) on MI455X / gfx1250, bf16 WMMA pipeline.
//   q = x@Wq^T+bq ; k = x@Wk^T+bk ; v = x@Wv^T+bv
//   num = exp(k+w) @ v ; den = exp(w) @ exp(k)
//   y   = sigmoid(q)*num/den ;  out = y@Wo^T+bo
// GEMMs: v_wmma_f32_16x16x32_bf16, double-buffered LDS staged with
// global_load_async_to_lds_b128 (ASYNCcnt) where the layout allows.
// ---------------------------------------------------------------------------

typedef __bf16 v16bf __attribute__((ext_vector_type(16)));
typedef float  v8f   __attribute__((ext_vector_type(8)));

struct FragBits { uint4 lo, hi; };   // 32 bytes == v16bf

__device__ __forceinline__ v16bf make_frag(uint4 lo, uint4 hi) {
  FragBits t{lo, hi};
  return __builtin_bit_cast(v16bf, t);
}

__device__ __forceinline__ unsigned short f2bf(float f) {
  union { float f; unsigned u; } x;
  x.f = f;
  unsigned u = x.u;
  unsigned r = u + 0x7FFFu + ((u >> 16) & 1u);   // round-to-nearest-even
  return (unsigned short)(r >> 16);
}

// 16B global -> LDS async copy (no VGPR round trip), tracked by ASYNCcnt.
__device__ __forceinline__ void async_cp16(const void* g, unsigned lds_addr) {
  asm volatile("global_load_async_to_lds_b128 %0, %1, off"
               :: "v"(lds_addr), "v"(g)
               : "memory");
}
__device__ __forceinline__ void wait_async0() {
  asm volatile("s_wait_asynccnt 0x0" ::: "memory");
}
// low 32 bits of a generic __shared__ pointer == LDS byte offset
__device__ __forceinline__ unsigned lds_off(const void* p) {
  return (unsigned)(unsigned long long)p;
}

// --------------------------- elementwise converts ---------------------------

__global__ void cvt_bf16_kernel(const float* __restrict__ s,
                                unsigned short* __restrict__ d, long long n) {
  long long i = (long long)blockIdx.x * blockDim.x + threadIdx.x;
  long long st = (long long)gridDim.x * blockDim.x;
  for (; i < n; i += st) d[i] = f2bf(s[i]);
}

__global__ void exp_bf16_kernel(const float* __restrict__ s,
                                unsigned short* __restrict__ d, long long n) {
  long long i = (long long)blockIdx.x * blockDim.x + threadIdx.x;
  long long st = (long long)gridDim.x * blockDim.x;
  for (; i < n; i += st) d[i] = f2bf(expf(s[i]));
}

// ------------------------------- WMMA GEMM ---------------------------------
// C[M,N] = A[M,K](bf16,row-major) @ B, where:
//   bt==0 : Bmat is [N][K] row-major (operand is Bmat^T) -> direct async stage
//   bt==1 : Bmat is [K][N] row-major                     -> transpose scatter
// 128x128 block tile, 256 threads (8 waves), K chunk = 64, double-buffered.

#define TM  128
#define TN  128
#define TKC 64

__global__ void __launch_bounds__(256)
wmma_gemm_kernel(const unsigned short* __restrict__ A, long long strideA, int lda,
                 const unsigned short* __restrict__ Bmat, long long strideB, int ldb, int bt,
                 int M, int N, int K, int mode,
                 const float* __restrict__ bias,
                 const float* __restrict__ wmat,
                 const float* __restrict__ sigq,
                 const float* __restrict__ numbuf, long long strideAux,
                 float* __restrict__ outF, long long strideOutF,
                 unsigned short* __restrict__ outB0, long long strideOutB0,
                 unsigned short* __restrict__ outB1,
                 int S) {
  (void)M;
  __shared__ unsigned short lsA[2][TM * TKC];   // 2 x 16 KiB
  __shared__ unsigned short lsB[2][TN * TKC];   // 2 x 16 KiB, stored [n][k]

  const int tid   = threadIdx.x;
  const int lane  = tid & 31;
  const int wv    = tid >> 5;
  const int waveM = wv >> 2;     // 0..1  -> 64-row band
  const int waveN = wv & 3;      // 0..3  -> 32-col band
  const int z     = blockIdx.z;

  A    += (long long)z * strideA;
  Bmat += (long long)z * strideB;

  const int rowBase = blockIdx.x * TM;
  const int colBase = blockIdx.y * TN;

  v8f acc[4][2];
#pragma unroll
  for (int mt = 0; mt < 4; ++mt)
#pragma unroll
    for (int nt = 0; nt < 2; ++nt)
#pragma unroll
      for (int r = 0; r < 8; ++r) acc[mt][nt][r] = 0.0f;

  // ---- stage one K-chunk (TM/TN x TKC) into LDS buffer `buf` ----
  auto stage = [&](int buf, int kc) {
    // A tile: 128x64 bf16 = 1024 16B chunks, async copy
#pragma unroll
    for (int it = 0; it < 4; ++it) {
      int c = tid + it * 256;
      int r = c >> 3, seg = (c & 7) * 8;
      async_cp16(A + (long long)(rowBase + r) * lda + kc + seg,
                 lds_off(&lsA[buf][r * TKC + seg]));
    }
    if (!bt) {
      // Bmat[N][K]: row n is the K-vector for column n, async copy
#pragma unroll
      for (int it = 0; it < 4; ++it) {
        int c = tid + it * 256;
        int r = c >> 3, seg = (c & 7) * 8;
        async_cp16(Bmat + (long long)(colBase + r) * ldb + kc + seg,
                   lds_off(&lsB[buf][r * TKC + seg]));
      }
    } else {
      // Bmat[K][N]: coalesced global read, scattered 16-bit LDS writes
#pragma unroll
      for (int it = 0; it < 4; ++it) {
        int c = tid + it * 256;
        int kk = c >> 4, n8 = (c & 15) * 8;
        uint4 d = *reinterpret_cast<const uint4*>(
            Bmat + (long long)(kc + kk) * ldb + colBase + n8);
        const unsigned short* e = reinterpret_cast<const unsigned short*>(&d);
#pragma unroll
        for (int i2 = 0; i2 < 8; ++i2) lsB[buf][(n8 + i2) * TKC + kk] = e[i2];
      }
    }
  };

  // ---- 16 WMMAs on buffer `buf` (two K=32 sub-steps) ----
  auto compute = [&](int buf) {
#pragma unroll
    for (int ks = 0; ks < TKC; ks += 32) {
      v16bf afrag[4], bfrag[2];
      const int off0 = ks + ((lane < 16) ? 0 : 8);   // A: K{off0..+7},{off0+16..+23}
#pragma unroll
      for (int mt = 0; mt < 4; ++mt) {
        int r = waveM * 64 + mt * 16 + (lane & 15);
        uint4 lo = *reinterpret_cast<const uint4*>(&lsA[buf][r * TKC + off0]);
        uint4 hi = *reinterpret_cast<const uint4*>(&lsA[buf][r * TKC + off0 + 16]);
        afrag[mt] = make_frag(lo, hi);
      }
      const int koff = ks + ((lane < 16) ? 0 : 8);   // B: K{koff..+7},{koff+16..+23}
#pragma unroll
      for (int nt = 0; nt < 2; ++nt) {
        int n = waveN * 32 + nt * 16 + (lane & 15);
        uint4 lo = *reinterpret_cast<const uint4*>(&lsB[buf][n * TKC + koff]);
        uint4 hi = *reinterpret_cast<const uint4*>(&lsB[buf][n * TKC + koff + 16]);
        bfrag[nt] = make_frag(lo, hi);
      }
#pragma unroll
      for (int mt = 0; mt < 4; ++mt)
#pragma unroll
        for (int nt = 0; nt < 2; ++nt)
          acc[mt][nt] = __builtin_amdgcn_wmma_f32_16x16x32_bf16(
              false, afrag[mt], false, bfrag[nt], (short)0, acc[mt][nt],
              false, false);
    }
  };

  // ---- double-buffered main loop ----
  const int nch = K / TKC;
  stage(0, 0);
  for (int i = 0; i < nch; ++i) {
    int cur = i & 1;
    wait_async0();          // my async copies into `cur` have landed
    __syncthreads();        // everyone's copies landed; prev readers of cur^1 done
    if (i + 1 < nch) stage(cur ^ 1, (i + 1) * TKC);
    compute(cur);
    __syncthreads();        // all reads of `cur` done before it is re-staged
  }

  // ---- epilogue (C/D: lane<16 -> M=r, lane>=16 -> M=8+r; N=lane%16) ----
  const int mAdd = (lane < 16) ? 0 : 8;
#pragma unroll
  for (int mt = 0; mt < 4; ++mt) {
#pragma unroll
    for (int nt = 0; nt < 2; ++nt) {
#pragma unroll
      for (int r = 0; r < 8; ++r) {
        int row = rowBase + waveM * 64 + mt * 16 + r + mAdd;
        int col = colBase + waveN * 32 + nt * 16 + (lane & 15);
        float a = acc[mt][nt][r];
        long long idx = (long long)row * N + col;
        switch (mode) {
          case 0: {  // Q: sigmoid(q) -> f32
            float t = a + bias[col];
            outF[idx] = 1.0f / (1.0f + expf(-t));
          } break;
          case 1: {  // K: exp(k) and exp(k + w[s,:]) -> bf16
            float t = a + bias[col];
            outB0[idx] = f2bf(expf(t));
            int s = row % S;
            outB1[idx] = f2bf(expf(t + wmat[(long long)s * N + col]));
          } break;
          case 2:    // V -> bf16
            outB0[idx] = f2bf(a + bias[col]);
            break;
          case 3:    // num -> f32 (batched)
            outF[(long long)z * strideOutF + idx] = a;
            break;
          case 4: {  // den; fuse y = sigq*num/den -> bf16 (batched)
            long long p = (long long)z * strideAux + idx;
            outB0[(long long)z * strideOutB0 + idx] =
                f2bf(sigq[p] * numbuf[p] / a);
          } break;
          case 5:    // final projection + bias -> f32 d_out
            outF[idx] = a + bias[col];
            break;
        }
      }
    }
  }
}

// ------------------------------- launcher ----------------------------------

extern "C" void kernel_launch(void* const* d_in, const int* in_sizes, int n_in,
                              void* d_out, int out_size, void* d_ws, size_t ws_size,
                              hipStream_t stream) {
  (void)in_sizes; (void)n_in; (void)out_size; (void)ws_size;
  const float* x     = (const float*)d_in[0];
  const float* wq_w  = (const float*)d_in[1];
  const float* wq_b  = (const float*)d_in[2];
  const float* wk_w  = (const float*)d_in[3];
  const float* wk_b  = (const float*)d_in[4];
  const float* wv_w  = (const float*)d_in[5];
  const float* wv_b  = (const float*)d_in[6];
  const float* wmat  = (const float*)d_in[7];
  const float* out_w = (const float*)d_in[8];
  const float* out_b = (const float*)d_in[9];

  const int Bn = 8, S = 1024, D = 1024;
  const long long MS = (long long)Bn * S;      // 8192 rows

  // workspace partitions (256B aligned)
  char* ws = (char*)d_ws;
  auto take = [&](size_t bytes) {
    char* p = ws;
    ws += (bytes + 255) & ~(size_t)255;
    return p;
  };
  unsigned short* xb    = (unsigned short*)take(MS * D * 2);        // x bf16
  unsigned short* wqb   = (unsigned short*)take((size_t)D * D * 2);
  unsigned short* wkb   = (unsigned short*)take((size_t)D * D * 2);
  unsigned short* wvb   = (unsigned short*)take((size_t)D * D * 2);
  unsigned short* owb   = (unsigned short*)take((size_t)D * D * 2);
  unsigned short* expw  = (unsigned short*)take((size_t)S * S * 2); // exp(w) bf16
  float*          sigq  = (float*)take(MS * D * 4);                 // sigmoid(q)
  unsigned short* expk  = (unsigned short*)take(MS * D * 2);        // exp(k) bf16
  unsigned short* expkw = (unsigned short*)take(MS * S * 2);        // exp(k+w) bf16
  unsigned short* vbb   = (unsigned short*)take(MS * D * 2);        // v bf16
  float*          numb  = (float*)take(MS * D * 4);                 // num f32
  unsigned short* yb    = (unsigned short*)take(MS * D * 2);        // y bf16

  // precision downconverts
  cvt_bf16_kernel<<<2048, 256, 0, stream>>>(x, xb, MS * D);
  cvt_bf16_kernel<<<512, 256, 0, stream>>>(wq_w, wqb, (long long)D * D);
  cvt_bf16_kernel<<<512, 256, 0, stream>>>(wk_w, wkb, (long long)D * D);
  cvt_bf16_kernel<<<512, 256, 0, stream>>>(wv_w, wvb, (long long)D * D);
  cvt_bf16_kernel<<<512, 256, 0, stream>>>(out_w, owb, (long long)D * D);
  exp_bf16_kernel<<<512, 256, 0, stream>>>(wmat, expw, (long long)S * S);

  dim3 blk(256);
  dim3 gProj(8192 / TM, D / TN, 1);   // (64, 8, 1)
  dim3 gAttn(S / TM, D / TN, Bn);     // (8, 8, 8)

  // Q projection -> sigmoid(q)
  wmma_gemm_kernel<<<gProj, blk, 0, stream>>>(
      xb, 0, D, wqb, 0, D, /*bt=*/0, (int)MS, D, D, /*mode=*/0,
      wq_b, nullptr, nullptr, nullptr, 0,
      sigq, 0, nullptr, 0, nullptr, S);

  // K projection -> exp(k), exp(k+w)
  wmma_gemm_kernel<<<gProj, blk, 0, stream>>>(
      xb, 0, D, wkb, 0, D, /*bt=*/0, (int)MS, D, D, /*mode=*/1,
      wk_b, wmat, nullptr, nullptr, 0,
      nullptr, 0, expk, 0, expkw, S);

  // V projection -> v bf16
  wmma_gemm_kernel<<<gProj, blk, 0, stream>>>(
      xb, 0, D, wvb, 0, D, /*bt=*/0, (int)MS, D, D, /*mode=*/2,
      wv_b, nullptr, nullptr, nullptr, 0,
      nullptr, 0, vbb, 0, nullptr, S);

  // num[b] = exp(k+w)[b] @ v[b]   (v is [K][N] -> transpose stage)
  wmma_gemm_kernel<<<gAttn, blk, 0, stream>>>(
      expkw, (long long)S * S, S, vbb, (long long)S * D, D, /*bt=*/1,
      S, D, S, /*mode=*/3,
      nullptr, nullptr, nullptr, nullptr, 0,
      numb, (long long)S * D, nullptr, 0, nullptr, S);

  // den[b] = exp(w) @ exp(k)[b] ; fuse y = sigq*num/den
  wmma_gemm_kernel<<<gAttn, blk, 0, stream>>>(
      expw, 0, S, expk, (long long)S * D, D, /*bt=*/1,
      S, D, S, /*mode=*/4,
      nullptr, nullptr, sigq, numb, (long long)S * D,
      nullptr, 0, yb, (long long)S * D, nullptr, S);

  // out = y @ out_w^T + out_b  -> f32 d_out
  wmma_gemm_kernel<<<gProj, blk, 0, stream>>>(
      yb, 0, D, owb, 0, D, /*bt=*/0, (int)MS, D, D, /*mode=*/5,
      out_b, nullptr, nullptr, nullptr, 0,
      (float*)d_out, 0, nullptr, 0, nullptr, S);
}
